// Ultimus_11536282157794
// MI455X (gfx1250) — matrix operand order repaired and checked by reading, not compiled, not run on hardware
//
#include <hip/hip_runtime.h>
#include <hip/hip_bf16.h>

typedef __attribute__((ext_vector_type(16))) _Float16 v16h;
typedef __attribute__((ext_vector_type(8)))  _Float16 v8h;
typedef __attribute__((ext_vector_type(8)))  float    v8f;

#define N_ROWS 8192
#define DIN    48
#define DATT   8

// ---- DPP helpers: cross-lane ops within a 16-lane DPP row (pure VALU) ------
template <int CTRL>
__device__ __forceinline__ float dpp_movf(float x) {
    int xi = __builtin_bit_cast(int, x);
    int r  = __builtin_amdgcn_update_dpp(xi, xi, CTRL, 0xF, 0xF, true);
    return __builtin_bit_cast(float, r);
}
__device__ __forceinline__ float rowmax16(float x) {
    x = fmaxf(x, dpp_movf<0xB1>(x));    // quad_perm [1,0,3,2]  : xor 1
    x = fmaxf(x, dpp_movf<0x4E>(x));    // quad_perm [2,3,0,1]  : xor 2
    x = fmaxf(x, dpp_movf<0x141>(x));   // row_half_mirror      : xor 4
    x = fmaxf(x, dpp_movf<0x140>(x));   // row_mirror           : xor 8
    return x;
}
__device__ __forceinline__ float rowsum16(float x) {
    x += dpp_movf<0xB1>(x);
    x += dpp_movf<0x4E>(x);
    x += dpp_movf<0x141>(x);
    x += dpp_movf<0x140>(x);
    return x;
}

// ---------------------------------------------------------------------------
// Kernel 1: QKV projections.  x[N,48] -> Xk,Xq (f16 [N,8]) and XvT (f16 [8,N]).
// Xk is pre-scaled by log2(e)/sqrt(8): softmax then needs only exp2.
// XvT is stored transposed so the P@V WMMA B-operand is a contiguous load.
// ---------------------------------------------------------------------------
__global__ __launch_bounds__(256) void ultimus_proj(
    const float* __restrict__ x,
    const float* __restrict__ Kw, const float* __restrict__ Kb,
    const float* __restrict__ Qw, const float* __restrict__ Qb,
    const float* __restrict__ Vw, const float* __restrict__ Vb,
    _Float16* __restrict__ Xk, _Float16* __restrict__ Xq,
    _Float16* __restrict__ XvT)
{
    __shared__ float wK[DIN * DATT];
    __shared__ float wQ[DIN * DATT];
    __shared__ float wV[DIN * DATT];
    __shared__ float bb[3 * DATT];

    for (int t = threadIdx.x; t < DIN * DATT; t += 256) {
        wK[t] = Kw[t]; wQ[t] = Qw[t]; wV[t] = Vw[t];
    }
    if (threadIdx.x < DATT) {
        bb[threadIdx.x]            = Kb[threadIdx.x];
        bb[DATT + threadIdx.x]     = Qb[threadIdx.x];
        bb[2 * DATT + threadIdx.x] = Vb[threadIdx.x];
    }
    __syncthreads();

    const int i = blockIdx.x * 256 + threadIdx.x;
    const float* xr = x + (size_t)i * DIN;

    float k[8], q[8], v[8];
    #pragma unroll
    for (int a = 0; a < 8; ++a) {
        k[a] = bb[a]; q[a] = bb[DATT + a]; v[a] = bb[2 * DATT + a];
    }
    #pragma unroll 4
    for (int c = 0; c < DIN; ++c) {
        float xc = xr[c];
        #pragma unroll
        for (int a = 0; a < 8; ++a) {
            k[a] = fmaf(xc, wK[c * 8 + a], k[a]);
            q[a] = fmaf(xc, wQ[c * 8 + a], q[a]);
            v[a] = fmaf(xc, wV[c * 8 + a], v[a]);
        }
    }

    // 1/sqrt(8) * log2(e): logits come out in the exp2 domain.
    const float kscale = 0.35355339059327373f * 1.4426950408889634f;
    v8h kh = {}, qh = {};
    #pragma unroll
    for (int a = 0; a < 8; ++a) {
        kh[a] = (_Float16)(k[a] * kscale);
        qh[a] = (_Float16)q[a];
        XvT[(size_t)a * N_ROWS + i] = (_Float16)v[a];
    }
    *(v8h*)(Xk + (size_t)i * 8) = kh;
    *(v8h*)(Xq + (size_t)i * 8) = qh;
}

// ---------------------------------------------------------------------------
// Kernel 2: fused flash attention, all-WMMA, 64 keys per iteration.
// 128 threads = 4 waves; each wave owns 16 query rows.  Per iteration:
//   4x v_wmma  S = Xk*Xq^T  (K padded 8->32),
//   online softmax (DPP row-max, exp2, DPP-free partial sums),
//   LDS round-trip of P (16x64 f16 tile) into A-layout,
//   2x v_wmma  acc = P*V + rescaled acc   (K = 32 keys each).
// ---------------------------------------------------------------------------
__global__ __launch_bounds__(128) void ultimus_attn(
    const _Float16* __restrict__ Xk, const _Float16* __restrict__ Xq,
    const _Float16* __restrict__ XvT, float* __restrict__ Z)
{
    // per-wave 16x64 f16 P tile; row stride 72 halves (144B: 16B-aligned, bank skew)
    __shared__ __align__(16) _Float16 pTile[4][16 * 72 + 8];

    const int lane = threadIdx.x & 31;
    const int wave = threadIdx.x >> 5;
    const int hi   = lane >> 4;      // 0: lanes 0-15, 1: lanes 16-31
    const int co   = lane & 15;
    const int i0   = (blockIdx.x * 4 + wave) * 16;

    _Float16* tp = &pTile[wave][0];

    const v8h  zero8  = {};
    const v16h zero16 = {};

    // A operand (Xk rows i0..i0+15): lanes 0-15 hold M=co, K=0..7 real, rest 0.
    v16h aK = zero16;
    if (hi == 0) {
        v8h t = *(const v8h*)(Xk + (size_t)(i0 + co) * DATT);
        aK = __builtin_shufflevector(t, zero8,
                                     0,1,2,3,4,5,6,7,8,9,10,11,12,13,14,15);
    }

    v8f acc = {};
    float m[8], lsum[8];
    #pragma unroll
    for (int r = 0; r < 8; ++r) { m[r] = -1e30f; lsum[r] = 0.0f; }

    for (int j0 = 0; j0 < N_ROWS; j0 += 64) {
        // Prefetch next chunk (L2-resident; emits global_prefetch_b8).
        if (j0 + 64 < N_ROWS) {
            __builtin_prefetch(Xq + (size_t)(j0 + 64 + lane) * DATT, 0, 1);
        }

        // B operands for S: Xq rows j0..j0+63 (four 16-col tiles), K=0..7 real.
        v16h bQ0 = zero16, bQ1 = zero16, bQ2 = zero16, bQ3 = zero16;
        if (hi == 0) {
            v8h q0 = *(const v8h*)(Xq + (size_t)(j0 +      co) * DATT);
            v8h q1 = *(const v8h*)(Xq + (size_t)(j0 + 16 + co) * DATT);
            v8h q2 = *(const v8h*)(Xq + (size_t)(j0 + 32 + co) * DATT);
            v8h q3 = *(const v8h*)(Xq + (size_t)(j0 + 48 + co) * DATT);
            bQ0 = __builtin_shufflevector(q0, zero8, 0,1,2,3,4,5,6,7,8,9,10,11,12,13,14,15);
            bQ1 = __builtin_shufflevector(q1, zero8, 0,1,2,3,4,5,6,7,8,9,10,11,12,13,14,15);
            bQ2 = __builtin_shufflevector(q2, zero8, 0,1,2,3,4,5,6,7,8,9,10,11,12,13,14,15);
            bQ3 = __builtin_shufflevector(q3, zero8, 0,1,2,3,4,5,6,7,8,9,10,11,12,13,14,15);
        }
        // B operands for P@V: column d=co of Xv, K = 32 keys per tile.
        v16h bV0 = zero16, bV1 = zero16;
        if (co < 8) {
            const _Float16* vp = XvT + (size_t)co * N_ROWS + j0 + hi * 8;
            bV0 = __builtin_shufflevector(*(const v8h*)(vp),
                                          *(const v8h*)(vp + 16),
                                          0,1,2,3,4,5,6,7,8,9,10,11,12,13,14,15);
            bV1 = __builtin_shufflevector(*(const v8h*)(vp + 32),
                                          *(const v8h*)(vp + 48),
                                          0,1,2,3,4,5,6,7,8,9,10,11,12,13,14,15);
        }

        v8f zc = {};
        v8f s0 = __builtin_amdgcn_wmma_f32_16x16x32_f16(false, aK, false, bQ0, (short)0, zc, false, false);
        v8f s1 = __builtin_amdgcn_wmma_f32_16x16x32_f16(false, aK, false, bQ1, (short)0, zc, false, false);
        v8f s2 = __builtin_amdgcn_wmma_f32_16x16x32_f16(false, aK, false, bQ2, (short)0, zc, false, false);
        v8f s3 = __builtin_amdgcn_wmma_f32_16x16x32_f16(false, aK, false, bQ3, (short)0, zc, false, false);

        // Online softmax (exp2 domain): lane holds s[row=r+8*hi][col=co] per VGPR r.
        v8f accC = {};
        #pragma unroll
        for (int r = 0; r < 8; ++r) {
            float bm = rowmax16(fmaxf(fmaxf(s0[r], s1[r]), fmaxf(s2[r], s3[r])));
            float mn = fmaxf(m[r], bm);
            float fr = exp2f(m[r] - mn);
            float p0 = exp2f(s0[r] - mn);
            float p1 = exp2f(s1[r] - mn);
            float p2 = exp2f(s2[r] - mn);
            float p3 = exp2f(s3[r] - mn);
            lsum[r] = lsum[r] * fr + ((p0 + p1) + (p2 + p3));
            m[r]    = mn;
            accC[r] = acc[r] * fr;          // rescale running accumulator (same C/D layout)
            const int row = r + 8 * hi;
            tp[row * 72 + co]      = (_Float16)p0;
            tp[row * 72 + co + 16] = (_Float16)p1;
            tp[row * 72 + co + 32] = (_Float16)p2;
            tp[row * 72 + co + 48] = (_Float16)p3;
        }

        // Wave-private tile; CDNA5 LDS ops are in-order within a wave, so a
        // compiler-level fence is sufficient (no workgroup barrier needed).
        __builtin_amdgcn_wave_barrier();
        asm volatile("" ::: "memory");

        // Re-read P in f16 A-layout: lane needs row=co, K = hi*8 + {0..7, 16..23}.
        {
            const _Float16* pr = tp + co * 72 + hi * 8;
            v16h pA0 = __builtin_shufflevector(*(const v8h*)(pr),
                                               *(const v8h*)(pr + 16),
                                               0,1,2,3,4,5,6,7,8,9,10,11,12,13,14,15);
            v16h pA1 = __builtin_shufflevector(*(const v8h*)(pr + 32),
                                               *(const v8h*)(pr + 48),
                                               0,1,2,3,4,5,6,7,8,9,10,11,12,13,14,15);
            acc = __builtin_amdgcn_wmma_f32_16x16x32_f16(false, pA0, false, bV0, (short)0, accC, false, false);
            acc = __builtin_amdgcn_wmma_f32_16x16x32_f16(false, pA1, false, bV1, (short)0, acc,  false, false);
        }

        asm volatile("" ::: "memory");
        __builtin_amdgcn_wave_barrier();
    }

    // Epilogue: DPP row-sum of l, normalize, write Z[N,8].
    #pragma unroll
    for (int r = 0; r < 8; ++r) {
        float l = rowsum16(lsum[r]);
        if (co < 8) {
            Z[(size_t)(i0 + r + 8 * hi) * DATT + co] = acc[r] / l;
        }
    }
}

// ---------------------------------------------------------------------------
// Kernel 3: residual + output projection: out = x + Z @ Zw + Zb.
// ---------------------------------------------------------------------------
__global__ __launch_bounds__(256) void ultimus_out(
    const float* __restrict__ x, const float* __restrict__ Zw,
    const float* __restrict__ Zb, const float* __restrict__ Zv,
    float* __restrict__ out)
{
    __shared__ float w[DATT * DIN];
    __shared__ float b[DIN];
    for (int t = threadIdx.x; t < DATT * DIN; t += 256) w[t] = Zw[t];
    if (threadIdx.x < DIN) b[threadIdx.x] = Zb[threadIdx.x];
    __syncthreads();

    const int i = blockIdx.x * 256 + threadIdx.x;
    float z[8];
    #pragma unroll
    for (int a = 0; a < 8; ++a) z[a] = Zv[(size_t)i * 8 + a];

    const float* xr = x + (size_t)i * DIN;
    float* orow = out + (size_t)i * DIN;
    #pragma unroll 4
    for (int c = 0; c < DIN; ++c) {
        float s = xr[c] + b[c];
        #pragma unroll
        for (int a = 0; a < 8; ++a) s = fmaf(z[a], w[a * DIN + c], s);
        orow[c] = s;
    }
}

// ---------------------------------------------------------------------------
extern "C" void kernel_launch(void* const* d_in, const int* in_sizes, int n_in,
                              void* d_out, int out_size, void* d_ws, size_t ws_size,
                              hipStream_t stream)
{
    const float* x  = (const float*)d_in[0];
    const float* Kw = (const float*)d_in[1];
    const float* Kb = (const float*)d_in[2];
    const float* Qw = (const float*)d_in[3];
    const float* Qb = (const float*)d_in[4];
    const float* Vw = (const float*)d_in[5];
    const float* Vb = (const float*)d_in[6];
    const float* Zw = (const float*)d_in[7];
    const float* Zb = (const float*)d_in[8];
    float* out = (float*)d_out;

    // Workspace layout (all 16B aligned):
    //   [0,       128KB) Xk  f16 [N,8]   (pre-scaled by log2e/sqrt(8))
    //   [128KB,   256KB) Xq  f16 [N,8]
    //   [256KB,   384KB) XvT f16 [8,N]
    //   [384KB,   640KB) Z   f32 [N,8]
    char* ws = (char*)d_ws;
    _Float16* Xk  = (_Float16*)(ws);
    _Float16* Xq  = (_Float16*)(ws + 131072);
    _Float16* XvT = (_Float16*)(ws + 262144);
    float*    Zv  = (float*)   (ws + 393216);

    ultimus_proj<<<N_ROWS / 256, 256, 0, stream>>>(x, Kw, Kb, Qw, Qb, Vw, Vb,
                                                   Xk, Xq, XvT);
    ultimus_attn<<<N_ROWS / 64, 128, 0, stream>>>(Xk, Xq, XvT, Zv);
    ultimus_out<<<N_ROWS / 256, 256, 0, stream>>>(x, Zw, Zb, Zv, out);
}
